// MLP_44890998177889
// MI455X (gfx1250) — matrix operand order repaired and checked by reading, not compiled, not run on hardware
//
#include <hip/hip_runtime.h>

typedef _Float16 f16;
typedef __attribute__((ext_vector_type(16))) _Float16 v16h;
typedef __attribute__((ext_vector_type(8)))  _Float16 v8h;
typedef __attribute__((ext_vector_type(8)))  float    v8f;
typedef __attribute__((ext_vector_type(4)))  float    v4f;

constexpr int E_   = 8;
constexpr int T_   = 2048;
constexpr int DIN_ = 2048;
constexpr int DH_  = 4096;

constexpr int BM  = 64;    // token rows per block
constexpr int BN  = 128;   // output cols per block
constexpr int BK  = 32;    // K per stage == one WMMA K step
constexpr int LDK = 40;    // padded LDS row (80B: 16B-aligned, 20dw -> 16 bank groups)

union Frag { v16h v; v8h h[2]; };

// ---------------------------------------------------------------------------
// Kernel 1: hidden[e][t][h] = silu(x@w1) * (x@w2), f32 in -> f16 out (ws)
// ---------------------------------------------------------------------------
__global__ __launch_bounds__(256)
void gateup_kernel(const float* __restrict__ x, const float* __restrict__ w1,
                   const float* __restrict__ w2, f16* __restrict__ hidden)
{
    __shared__ __align__(16) f16 Xs [2][BM][LDK];   // 10 KB
    __shared__ __align__(16) f16 W1s[2][BN][LDK];   // 20 KB  ([n][k] = B^T)
    __shared__ __align__(16) f16 W2s[2][BN][LDK];   // 20 KB

    const int tid  = threadIdx.x;
    const int le   = blockIdx.z;            // local expert index
    const int bm   = blockIdx.x * BM;
    const int bn   = blockIdx.y * BN;
    const int lane = tid & 31;
    const int wave = tid >> 5;
    const int wr   = (wave & 1) * 32;       // wave row band
    const int wc   = (wave >> 1) * 32;      // wave col band
    const int lm   = lane & 15;
    const int lh   = lane >> 4;

    // global loader mapping (256 threads)
    const int xRow = tid >> 2;              // 0..63
    const int xK   = (tid & 3) * 8;         // 0,8,16,24
    const int wN   = tid & 127;             // lane-contiguous -> coalesced
    const int wK   = (tid >> 7) * 16;       // 0 or 16

    const float* xg  = x  + (size_t)(le * T_ + bm + xRow) * DIN_ + xK;
    const float* w1g = w1 + ((size_t)le * DIN_ + wK) * DH_ + bn + wN;
    const float* w2g = w2 + ((size_t)le * DIN_ + wK) * DH_ + bn + wN;

    float xr[8], w1r[16], w2r[16];

    auto load_g = [&](int kk) {
        const int k0 = kk * BK;
        v4f a = *(const v4f*)(xg + k0);
        v4f b = *(const v4f*)(xg + k0 + 4);
        #pragma unroll
        for (int i = 0; i < 4; ++i) { xr[i] = a[i]; xr[4 + i] = b[i]; }
        #pragma unroll
        for (int i = 0; i < 16; ++i) {
            w1r[i] = w1g[(size_t)(k0 + i) * DH_];   // 128B/line per wave instr
            w2r[i] = w2g[(size_t)(k0 + i) * DH_];
        }
    };

    auto store_l = [&](int buf) {
        v8h xv, w1lo, w1hi, w2lo, w2hi;
        #pragma unroll
        for (int i = 0; i < 8; ++i) {
            xv[i]   = (f16)xr[i];
            w1lo[i] = (f16)w1r[i];  w1hi[i] = (f16)w1r[8 + i];
            w2lo[i] = (f16)w2r[i];  w2hi[i] = (f16)w2r[8 + i];
        }
        *(v8h*)&Xs [buf][xRow][xK]   = xv;
        *(v8h*)&W1s[buf][wN][wK]     = w1lo;
        *(v8h*)&W1s[buf][wN][wK + 8] = w1hi;
        *(v8h*)&W2s[buf][wN][wK]     = w2lo;
        *(v8h*)&W2s[buf][wN][wK + 8] = w2hi;
    };

    v8f gacc[2][2] = {};
    v8f uacc[2][2] = {};

    auto compute = [&](int buf) {
        Frag a[2], b1[2], b2[2];
        #pragma unroll
        for (int i = 0; i < 2; ++i) {
            // A 16x32 f16, contiguous k order chosen here (B matches): lane = row M, K = lh*16 .. lh*16+15
            const f16* p = &Xs[buf][wr + i * 16 + lm][0];
            a[i].h[0] = *(const v8h*)(p + lh * 16);
            a[i].h[1] = *(const v8h*)(p + lh * 16 + 8);
        }
        #pragma unroll
        for (int j = 0; j < 2; ++j) {
            // B 32x16 f16, same contiguous k order as A: lane = col N, K = lh*16 .. lh*16+15
            const f16* p1 = &W1s[buf][wc + j * 16 + lm][0];
            const f16* p2 = &W2s[buf][wc + j * 16 + lm][0];
            b1[j].h[0] = *(const v8h*)(p1 + lh * 16);
            b1[j].h[1] = *(const v8h*)(p1 + lh * 16 + 8);
            b2[j].h[0] = *(const v8h*)(p2 + lh * 16);
            b2[j].h[1] = *(const v8h*)(p2 + lh * 16 + 8);
        }
        #pragma unroll
        for (int i = 0; i < 2; ++i)
            #pragma unroll
            for (int j = 0; j < 2; ++j) {
                gacc[i][j] = __builtin_amdgcn_wmma_f32_16x16x32_f16(
                    false, a[i].v, false, b1[j].v, (short)0, gacc[i][j], false, false);
                uacc[i][j] = __builtin_amdgcn_wmma_f32_16x16x32_f16(
                    false, a[i].v, false, b2[j].v, (short)0, uacc[i][j], false, false);
            }
    };

    constexpr int NSTEP = DIN_ / BK;   // 64
    load_g(0);
    store_l(0);
    load_g(1);
    // single-barrier pipeline: barrier(k) publishes stage-k stores AND orders
    // stage-(k-1) readers of the buffer we are about to overwrite.
    for (int kk = 0; kk < NSTEP; ++kk) {
        __syncthreads();
        if (kk + 1 < NSTEP) store_l((kk + 1) & 1);
        if (kk + 2 < NSTEP) load_g(kk + 2);
        compute(kk & 1);
    }

    // SwiGLU epilogue. C/D layout: lane<16 -> M=r, lane>=16 -> M=r+8, N=lane%16
    #pragma unroll
    for (int i = 0; i < 2; ++i)
        #pragma unroll
        for (int j = 0; j < 2; ++j)
            #pragma unroll
            for (int r = 0; r < 8; ++r) {
                float g = gacc[i][j][r];
                float u = uacc[i][j][r];
                float h = (g / (1.0f + __expf(-g))) * u;   // silu(g) * u
                int row = bm + wr + i * 16 + r + lh * 8;
                int col = bn + wc + j * 16 + lm;
                hidden[(size_t)(le * T_ + row) * DH_ + col] = (f16)h;
            }
}

// ---------------------------------------------------------------------------
// Kernel 2: out[e][t][d] = hidden(f16) @ w3(f32->f16), f32 out
// ---------------------------------------------------------------------------
__global__ __launch_bounds__(256)
void down_kernel(const f16* __restrict__ hidden, const float* __restrict__ w3,
                 float* __restrict__ out)
{
    __shared__ __align__(16) f16 Hs [2][BM][LDK];   // 10 KB
    __shared__ __align__(16) f16 W3s[2][BN][LDK];   // 20 KB

    const int tid  = threadIdx.x;
    const int le   = blockIdx.z;
    const int bm   = blockIdx.x * BM;
    const int bn   = blockIdx.y * BN;
    const int lane = tid & 31;
    const int wave = tid >> 5;
    const int wr   = (wave & 1) * 32;
    const int wc   = (wave >> 1) * 32;
    const int lm   = lane & 15;
    const int lh   = lane >> 4;

    const int xRow = tid >> 2;
    const int xK   = (tid & 3) * 8;
    const int wN   = tid & 127;
    const int wK   = (tid >> 7) * 16;

    const f16*   hg  = hidden + (size_t)(le * T_ + bm + xRow) * DH_ + xK;
    const float* w3g = w3 + ((size_t)le * DH_ + wK) * DIN_ + bn + wN;

    v8h   hr;
    float w3r[16];

    auto load_g = [&](int kk) {
        const int k0 = kk * BK;
        hr = *(const v8h*)(hg + k0);            // already f16
        #pragma unroll
        for (int i = 0; i < 16; ++i)
            w3r[i] = w3g[(size_t)(k0 + i) * DIN_];
    };

    auto store_l = [&](int buf) {
        v8h lo, hi;
        #pragma unroll
        for (int i = 0; i < 8; ++i) { lo[i] = (f16)w3r[i]; hi[i] = (f16)w3r[8 + i]; }
        *(v8h*)&Hs [buf][xRow][xK]   = hr;
        *(v8h*)&W3s[buf][wN][wK]     = lo;
        *(v8h*)&W3s[buf][wN][wK + 8] = hi;
    };

    v8f acc[2][2] = {};

    auto compute = [&](int buf) {
        Frag a[2], b[2];
        #pragma unroll
        for (int i = 0; i < 2; ++i) {
            const f16* p = &Hs[buf][wr + i * 16 + lm][0];
            a[i].h[0] = *(const v8h*)(p + lh * 16);
            a[i].h[1] = *(const v8h*)(p + lh * 16 + 8);
        }
        #pragma unroll
        for (int j = 0; j < 2; ++j) {
            const f16* p = &W3s[buf][wc + j * 16 + lm][0];
            b[j].h[0] = *(const v8h*)(p + lh * 16);
            b[j].h[1] = *(const v8h*)(p + lh * 16 + 8);
        }
        #pragma unroll
        for (int i = 0; i < 2; ++i)
            #pragma unroll
            for (int j = 0; j < 2; ++j)
                acc[i][j] = __builtin_amdgcn_wmma_f32_16x16x32_f16(
                    false, a[i].v, false, b[j].v, (short)0, acc[i][j], false, false);
    };

    constexpr int NSTEP = DH_ / BK;   // 128
    load_g(0);
    store_l(0);
    load_g(1);
    for (int kk = 0; kk < NSTEP; ++kk) {
        __syncthreads();
        if (kk + 1 < NSTEP) store_l((kk + 1) & 1);
        if (kk + 2 < NSTEP) load_g(kk + 2);
        compute(kk & 1);
    }

    #pragma unroll
    for (int i = 0; i < 2; ++i)
        #pragma unroll
        for (int j = 0; j < 2; ++j)
            #pragma unroll
            for (int r = 0; r < 8; ++r) {
                int row = bm + wr + i * 16 + r + lh * 8;
                int col = bn + wc + j * 16 + lm;
                out[(size_t)(le * T_ + row) * DIN_ + col] = acc[i][j][r];
            }
}

// ---------------------------------------------------------------------------
extern "C" void kernel_launch(void* const* d_in, const int* in_sizes, int n_in,
                              void* d_out, int out_size, void* d_ws, size_t ws_size,
                              hipStream_t stream) {
    const float* x  = (const float*)d_in[0];
    const float* w1 = (const float*)d_in[1];
    const float* w2 = (const float*)d_in[2];
    const float* w3 = (const float*)d_in[3];
    float* out  = (float*)d_out;
    f16* hidden = (f16*)d_ws;

    const size_t need_all = (size_t)E_ * T_ * DH_ * sizeof(f16);  // 128 MiB
    const int chunkE = (ws_size >= need_all) ? E_ : 1;            // else 16 MiB/expert

    dim3 block(256);
    for (int e0 = 0; e0 < E_; e0 += chunkE) {
        dim3 gA(T_ / BM, DH_ / BN, chunkE);
        gateup_kernel<<<gA, block, 0, stream>>>(
            x  + (size_t)e0 * T_ * DIN_,
            w1 + (size_t)e0 * DIN_ * DH_,
            w2 + (size_t)e0 * DIN_ * DH_,
            hidden);
        dim3 gB(T_ / BM, DIN_ / BN, chunkE);
        down_kernel<<<gB, block, 0, stream>>>(
            hidden,
            w3  + (size_t)e0 * DH_ * DIN_,
            out + (size_t)e0 * T_ * DIN_);
    }
}
